// LLaDA2MoeAttention_506806141419
// MI455X (gfx1250) — matrix operand order repaired and checked by reading, not compiled, not run on hardware
//
#include <hip/hip_runtime.h>

// ---------------------------------------------------------------------------
// LLaDA2 attention for MI455X (gfx1250): bf16 WMMA pipeline.
//   K1: QKV GEMM  (f32 in -> bf16 qkv in ws)        v_wmma_f32_16x16x32_bf16
//   K2: RMSNorm + RoPE on q,k rows (in-place bf16)  ds_swizzle reductions
//   K3: flash attention, block-causal @32           WMMA QK^T / PV
//       K tile staged by the Tensor Data Mover      tensor_load_to_lds
//   K4: output GEMM (bf16 attn-out @ f32 w_o -> f32 d_out)
// ---------------------------------------------------------------------------

typedef __bf16 bf16;
typedef __bf16 v16bf __attribute__((ext_vector_type(16)));
typedef __bf16 v8bf  __attribute__((ext_vector_type(8)));
typedef float  v8f   __attribute__((ext_vector_type(8)));
typedef float  v4f   __attribute__((ext_vector_type(4)));
typedef unsigned int v4u __attribute__((ext_vector_type(4)));
typedef int    v8i   __attribute__((ext_vector_type(8)));
typedef int    v4i   __attribute__((ext_vector_type(4)));

#define S_TOT   4096
#define HID     2048
#define NH      16
#define NKV     4
#define DH      128
#define NQKV    3072          // (16 + 2*4) * 128
#define KOFF    2048          // start of K in qkv row
#define VOFF    2560          // start of V in qkv row
#define ATT_SCALE 0.08838834764831845f
#define RMS_EPS 1e-6f

// cross-lane xor exchange via ds_swizzle (group-of-32: xor in [14:10], and=0x1F)
#define SWZ_F(v, m) __int_as_float(__builtin_amdgcn_ds_swizzle(__float_as_int(v), (((m) << 10) | 0x1F)))

__device__ __forceinline__ v16bf mk16(v8bf lo, v8bf hi) {
  return __builtin_shufflevector(lo, hi, 0, 1, 2, 3, 4, 5, 6, 7, 8, 9, 10, 11, 12, 13, 14, 15);
}

// A-fragment (16x32 bf16, MxK): lane = M | (kHalf<<4); elements 0..7 -> K = e + 8*kHalf,
// elements 8..15 -> K = e + 8 + 8*kHalf.  src points at row M (32 contiguous K values).
__device__ __forceinline__ v16bf load_a_frag(const bf16* row, int kHalf) {
  v8bf lo = *(const v8bf*)(row + 8 * kHalf);
  v8bf hi = *(const v8bf*)(row + 16 + 8 * kHalf);
  return mk16(lo, hi);
}

// B-fragment (32x16 bf16, KxN): lane = N | (kHalf<<4); elements 0..7 -> K = e + 8*kHalf,
// elements 8..15 -> K = e + 8 + 8*kHalf.
// src points at the 32 contiguous K values for column N (data stored transposed [n][k]).
__device__ __forceinline__ v16bf load_b_frag(const bf16* coln, int kHalf) {
  v8bf lo = *(const v8bf*)(coln + 8 * kHalf);
  v8bf hi = *(const v8bf*)(coln + 16 + 8 * kHalf);
  return mk16(lo, hi);
}

// ---- Tensor Data Mover: 2D tile of bf16 rows -> LDS (with row padding) -----
// tile_d0 elems/row (2B units), tile_d1 rows, row stride 'stride' elems,
// LDS pad: 16B appended every 256B (matches sK[32][136]).
__device__ __forceinline__ void tdm_load_tile_2d(unsigned lds_off,
                                                 unsigned long long gaddr,
                                                 int tile_d0, int tile_d1,
                                                 int stride) {
  v4u g0;
  g0[0] = 1u;                                        // count=1, user descriptor
  g0[1] = lds_off;                                   // lds_addr (bytes)
  g0[2] = (unsigned)(gaddr & 0xFFFFFFFFull);         // global_addr[31:0]
  g0[3] = (unsigned)((gaddr >> 32) & 0x1FFFFFFull)   // global_addr[56:32]
          | (2u << 30);                              // type = 2 (image)
  v8i g1;
  g1[0] = (1 << 16)                                  // data_size = 2 bytes
          | (1 << 20)                                // pad_enable
          | (5 << 22)                                // pad_interval: 64 DWORDs (256B)
          | (3 << 25);                               // pad_amount: 4 DWORDs (16B)
  g1[1] = (int)((unsigned)(stride & 0xFFFF) << 16);  // tensor_dim0[15:0]
  g1[2] = (int)(((unsigned)stride >> 16) | (((unsigned)S_TOT & 0xFFFF) << 16));
  g1[3] = (int)(((unsigned)S_TOT >> 16) | ((unsigned)tile_d0 << 16));
  g1[4] = tile_d1;                                   // tile_dim1 (tile_dim2 = 0)
  g1[5] = stride;                                    // tensor_dim0_stride[31:0]
  g1[6] = 0;
  g1[7] = 0;
  v4i z4 = {};
#if __clang_major__ >= 23
  v8i z8 = {};
  __builtin_amdgcn_tensor_load_to_lds(g0, g1, z4, z4, z8, 0);
#else
  __builtin_amdgcn_tensor_load_to_lds(g0, g1, z4, z4, 0);
#endif
}

// ---- load/store helpers for the templated GEMM --------------------------------
__device__ __forceinline__ void loadA16(bf16* dst, const float* src) {
  const v4f* s4 = (const v4f*)src;          // 16B-aligned by construction
#pragma unroll
  for (int i = 0; i < 4; ++i) {
    v4f t = s4[i];
    dst[4 * i + 0] = (bf16)t[0];
    dst[4 * i + 1] = (bf16)t[1];
    dst[4 * i + 2] = (bf16)t[2];
    dst[4 * i + 3] = (bf16)t[3];
  }
}
__device__ __forceinline__ void loadA16(bf16* dst, const bf16* src) {
  const v8bf* s8 = (const v8bf*)src;        // 16B-aligned by construction
  *(v8bf*)(dst) = s8[0];
  *(v8bf*)(dst + 8) = s8[1];
}
__device__ __forceinline__ void storeC(float* p, float v) { *p = v; }
__device__ __forceinline__ void storeC(bf16* p, float v)  { *p = (bf16)v; }

// ---------------------------------------------------------------------------
// Generic bf16-WMMA GEMM: C[M,N] = A[M,K] * B[K,N].  A is f32 or bf16, B is f32.
// Block: 256 threads (8 waves), 128x128 C tile, K step 32.
// Wave (wm in 0..3, wn in 0..1) owns a 32x64 sub-tile = 2x4 WMMA tiles.
// ---------------------------------------------------------------------------
template <typename AT, typename OT>
__global__ __launch_bounds__(256) void gemm_bf16_wmma(const AT* __restrict__ A,
                                                      const float* __restrict__ Bw,
                                                      OT* __restrict__ C,
                                                      int M, int N, int K) {
  __shared__ bf16 sA[128][40];   // [m][k], row stride 80B (16B aligned)
  __shared__ bf16 sB[128][40];   // [n][k] (transposed stage for contiguous B frags)

  const int tid = threadIdx.x;
  const int lane = tid & 31, wid = tid >> 5;
  const int wm = wid >> 1, wn = wid & 1;
  const int half = lane >> 4, l15 = lane & 15;
  const int row0 = blockIdx.y * 128, col0 = blockIdx.x * 128;

  v8f acc[2][4] = {};

  const int arow = tid >> 1, aseg = (tid & 1) * 16;   // A loader: 2 threads/row
  const int bkr = tid >> 3, bseg = (tid & 7) * 16;    // B loader: 8 threads/row

  for (int k0 = 0; k0 < K; k0 += 32) {
    // ---- stage A 128x32 (convert to bf16) ----
    loadA16(&sA[arow][aseg], A + (size_t)(row0 + arow) * K + k0 + aseg);
    if (k0 + 32 < K)
      __builtin_prefetch(A + (size_t)(row0 + arow) * K + k0 + 32 + aseg, 0, 1);
    // ---- stage B 32x128 transposed into sB[n][k] ----
    {
      const v4f* bp4 = (const v4f*)(Bw + (size_t)(k0 + bkr) * N + col0 + bseg);
#pragma unroll
      for (int i = 0; i < 4; ++i) {
        v4f t = bp4[i];
        sB[bseg + 4 * i + 0][bkr] = (bf16)t[0];
        sB[bseg + 4 * i + 1][bkr] = (bf16)t[1];
        sB[bseg + 4 * i + 2][bkr] = (bf16)t[2];
        sB[bseg + 4 * i + 3][bkr] = (bf16)t[3];
      }
      if (k0 + 32 < K)
        __builtin_prefetch(Bw + (size_t)(k0 + 32 + bkr) * N + col0 + bseg, 0, 1);
    }
    __syncthreads();

    v16bf af[2], bfr[4];
#pragma unroll
    for (int mt = 0; mt < 2; ++mt)
      af[mt] = load_a_frag(&sA[wm * 32 + mt * 16 + l15][0], half);
#pragma unroll
    for (int nt = 0; nt < 4; ++nt)
      bfr[nt] = load_b_frag(&sB[wn * 64 + nt * 16 + l15][0], half);

#pragma unroll
    for (int mt = 0; mt < 2; ++mt)
#pragma unroll
      for (int nt = 0; nt < 4; ++nt)
        acc[mt][nt] = __builtin_amdgcn_wmma_f32_16x16x32_bf16(
            false, af[mt], false, bfr[nt], (short)0, acc[mt][nt], false, false);
    __syncthreads();
  }

  // ---- epilogue: C/D layout is lane=N|half<<4, vgpr r -> M = r + 8*half ----
#pragma unroll
  for (int mt = 0; mt < 2; ++mt)
#pragma unroll
    for (int nt = 0; nt < 4; ++nt)
#pragma unroll
      for (int r = 0; r < 8; ++r) {
        int rr = row0 + wm * 32 + mt * 16 + r + 8 * half;
        int cc = col0 + wn * 64 + nt * 16 + l15;
        storeC(C + (size_t)rr * N + cc, acc[mt][nt][r]);
      }
}

// ---------------------------------------------------------------------------
// RMSNorm + RoPE over q (16 heads) and k (4 heads) rows of qkv, in place (bf16).
// One wave per 128-wide row; 4 elements/lane; wave32 reductions via ds_swizzle.
// ---------------------------------------------------------------------------
__global__ __launch_bounds__(256) void rmsnorm_rope_kernel(bf16* __restrict__ qkv,
                                                           const float* __restrict__ cosT,
                                                           const float* __restrict__ sinT,
                                                           const float* __restrict__ qw,
                                                           const float* __restrict__ kw) {
  const int lane = threadIdx.x & 31, wid = threadIdx.x >> 5;
  const int rid = blockIdx.x * 8 + wid;            // 0 .. S*(NH+NKV)-1
  const int s = rid / (NH + NKV);
  const int j = rid % (NH + NKV);
  const int col0 = (j < NH) ? j * DH : KOFF + (j - NH) * DH;
  const float* w = (j < NH) ? qw : kw;

  bf16* p = qkv + (size_t)s * NQKV + col0;
  const int d0 = lane * 4;

  float x[4], ss = 0.f;
#pragma unroll
  for (int i = 0; i < 4; ++i) { x[i] = (float)p[d0 + i]; ss += x[i] * x[i]; }
  ss += SWZ_F(ss, 1);
  ss += SWZ_F(ss, 2);
  ss += SWZ_F(ss, 4);
  ss += SWZ_F(ss, 8);
  ss += SWZ_F(ss, 16);
  const float rinv = rsqrtf(ss * (1.f / (float)DH) + RMS_EPS);

  float nv[4];
#pragma unroll
  for (int i = 0; i < 4; ++i) nv[i] = x[i] * rinv * w[d0 + i];

  const float* cp = cosT + (size_t)s * DH + d0;
  const float* sp = sinT + (size_t)s * DH + d0;
#pragma unroll
  for (int i = 0; i < 4; ++i) {
    float partner = SWZ_F(nv[i], 16);              // d <-> d +/- 64 is lane ^ 16
    float rot = (lane < 16) ? -partner : partner;  // rotate_half sign
    p[d0 + i] = (bf16)(nv[i] * cp[i] + rot * sp[i]);
  }
}

// ---------------------------------------------------------------------------
// Flash attention, block-causal (BLOCK=32).  Grid: (S/128, NH).
// 8 waves; wave wq owns 16 query rows [qt*128+wq*16, +16) — always inside one
// 32-block, so the mask reduces to a per-wave key-count (no element masking).
// K tile (32 keys x 128 bf16) staged by the Tensor Data Mover (wave 0 issues,
// TENSORcnt tracked); V tile staged manually (TDM cannot transpose).
// ---------------------------------------------------------------------------
__global__ __launch_bounds__(256) void flash_attn_kernel(const bf16* __restrict__ qkv,
                                                         bf16* __restrict__ aout) {
  __shared__ bf16 sK[32][136];      // [key][d]   row 272B = 256B data + 16B TDM pad
  __shared__ bf16 sVt[128][40];     // [d][key]   transposed for B-fragments
  __shared__ bf16 sP[8][16][40];    // per-wave P round-trip (C-layout -> A-layout)

  const int tid = threadIdx.x, lane = tid & 31, wq = tid >> 5;
  const int half = lane >> 4, l15 = lane & 15;
  const int qt = blockIdx.x, h = blockIdx.y;
  const int kvh = h >> 2;                    // GQA: 4 q-heads per kv-head
  const int qrow0 = qt * 128 + wq * 16;
  const int qtb = qrow0 >> 5;                // this wave's 32-block index

  const unsigned sK_lds_off = (unsigned)(unsigned long long)(void*)&sK[0][0];

  // Q fragments (16 rows x 128 d) straight from global, kept in registers.
  v16bf qf[4];
  {
    const bf16* qp = qkv + (size_t)(qrow0 + l15) * NQKV + h * DH;
#pragma unroll
    for (int c = 0; c < 4; ++c) qf[c] = load_a_frag(qp + c * 32, half);
  }

  float rmax[8], rsum[8];
#pragma unroll
  for (int r = 0; r < 8; ++r) { rmax[r] = -1e30f; rsum[r] = 0.f; }
  v8f o[8] = {};

  const int nkb = qt * 4 + 4;                // max 32-key blocks any wave needs
  const int kcol = KOFF + kvh * DH, vcol = VOFF + kvh * DH;
  const int lkey = tid >> 3, ldseg = (tid & 7) * 16;   // cooperative loader map

  for (int kb = 0; kb < nkb; ++kb) {
    // ---- K tile via TDM (issued once per workgroup, overlaps V staging) ----
    if (wq == 0) {
      unsigned long long ga =
          (unsigned long long)(const void*)(qkv + (size_t)(kb * 32) * NQKV + kcol);
      tdm_load_tile_2d(sK_lds_off, ga, DH, 32, NQKV);
    }
    // ---- V tile staged transposed into sVt[d][key] (bf16) ----
    {
      const bf16* vp = qkv + (size_t)(kb * 32 + lkey) * NQKV + vcol + ldseg;
      v8bf va = *(const v8bf*)(vp);
      v8bf vb = *(const v8bf*)(vp + 8);
#pragma unroll
      for (int i = 0; i < 8; ++i) {
        sVt[ldseg + i][lkey] = va[i];
        sVt[ldseg + 8 + i][lkey] = vb[i];
      }
      if (kb + 1 < nkb)
        __builtin_prefetch(qkv + (size_t)((kb + 1) * 32 + lkey) * NQKV + vcol + ldseg, 0, 1);
    }
    if (wq == 0) __builtin_amdgcn_s_wait_tensorcnt(0);  // K tile resident
    __syncthreads();

    if (kb <= qtb) {                         // block-causal: whole tile allowed
      // ---- S = Q K^T over d (4 WMMAs per 16-key half) ----
      v8f s0 = {}, s1 = {};
#pragma unroll
      for (int c = 0; c < 4; ++c) {
        v16bf bk0 = load_b_frag(&sK[l15][c * 32], half);        // keys 0..15
        v16bf bk1 = load_b_frag(&sK[16 + l15][c * 32], half);   // keys 16..31
        s0 = __builtin_amdgcn_wmma_f32_16x16x32_bf16(false, qf[c], false, bk0,
                                                     (short)0, s0, false, false);
        s1 = __builtin_amdgcn_wmma_f32_16x16x32_bf16(false, qf[c], false, bk1,
                                                     (short)0, s1, false, false);
      }
      // ---- online softmax (per M-row: reduce across the 16 lanes of a half) ----
#pragma unroll
      for (int r = 0; r < 8; ++r) {
        float a0 = s0[r] * ATT_SCALE, a1 = s1[r] * ATT_SCALE;
        float mx = fmaxf(a0, a1);
        mx = fmaxf(mx, SWZ_F(mx, 1));
        mx = fmaxf(mx, SWZ_F(mx, 2));
        mx = fmaxf(mx, SWZ_F(mx, 4));
        mx = fmaxf(mx, SWZ_F(mx, 8));
        float nmax = fmaxf(rmax[r], mx);
        float corr = __expf(rmax[r] - nmax);
        float p0 = __expf(a0 - nmax), p1 = __expf(a1 - nmax);
        float ts = p0 + p1;
        ts += SWZ_F(ts, 1);
        ts += SWZ_F(ts, 2);
        ts += SWZ_F(ts, 4);
        ts += SWZ_F(ts, 8);
        rsum[r] = rsum[r] * corr + ts;
        rmax[r] = nmax;
        s0[r] = p0;
        s1[r] = p1;
#pragma unroll
        for (int nt = 0; nt < 8; ++nt) o[nt][r] *= corr;
      }
      // ---- P: C-layout regs -> LDS -> A-layout fragment (wave-private) ----
#pragma unroll
      for (int r = 0; r < 8; ++r) {
        int m = r + 8 * half;
        sP[wq][m][l15] = (bf16)s0[r];
        sP[wq][m][16 + l15] = (bf16)s1[r];
      }
      asm volatile("s_wait_dscnt 0" ::: "memory");   // same-wave LDS RAW fence
      v16bf pf = load_a_frag(&sP[wq][l15][0], half);
      // ---- O += P V (8 d-tiles of 16) ----
#pragma unroll
      for (int nt = 0; nt < 8; ++nt) {
        v16bf bv = load_b_frag(&sVt[nt * 16 + l15][0], half);
        o[nt] = __builtin_amdgcn_wmma_f32_16x16x32_bf16(false, pf, false, bv,
                                                        (short)0, o[nt], false, false);
      }
    }
    __syncthreads();   // protects sK/sVt against next iteration's overwrite
  }

  // ---- normalize and write attention output (bf16, [s][h*128+d]) ----
#pragma unroll
  for (int nt = 0; nt < 8; ++nt)
#pragma unroll
    for (int r = 0; r < 8; ++r) {
      int srow = qrow0 + r + 8 * half;
      int cc = h * DH + nt * 16 + l15;
      aout[(size_t)srow * (NH * DH) + cc] = (bf16)(o[nt][r] / rsum[r]);
    }
}

// ---------------------------------------------------------------------------
extern "C" void kernel_launch(void* const* d_in, const int* in_sizes, int n_in,
                              void* d_out, int out_size, void* d_ws, size_t ws_size,
                              hipStream_t stream) {
  (void)in_sizes; (void)n_in; (void)out_size; (void)ws_size;
  const float* hidden = (const float*)d_in[0];
  const float* cosT   = (const float*)d_in[1];
  const float* sinT   = (const float*)d_in[2];
  const float* w_qkv  = (const float*)d_in[3];
  const float* q_norm = (const float*)d_in[4];
  const float* k_norm = (const float*)d_in[5];
  const float* w_o    = (const float*)d_in[6];
  float* out = (float*)d_out;

  bf16* qkv  = (bf16*)d_ws;                          // 4096*3072 bf16 = 25.2 MB
  bf16* aout = qkv + (size_t)S_TOT * NQKV;           // 4096*2048 bf16 = 16.8 MB

  // 1) QKV projection: f32 x f32 -> bf16 qkv
  gemm_bf16_wmma<float, bf16><<<dim3(NQKV / 128, S_TOT / 128), 256, 0, stream>>>(
      hidden, w_qkv, qkv, S_TOT, NQKV, HID);

  // 2) RMSNorm + RoPE on q,k rows (in place). S*(NH+NKV) rows, 8 rows/block.
  rmsnorm_rope_kernel<<<(S_TOT * (NH + NKV)) / 8, 256, 0, stream>>>(
      qkv, cosT, sinT, q_norm, k_norm);

  // 3) Block-causal flash attention -> bf16 attn-out (K tile via TDM)
  flash_attn_kernel<<<dim3(S_TOT / 128, NH), 256, 0, stream>>>(qkv, aout);

  // 4) Output projection: bf16 x f32 -> f32 d_out
  gemm_bf16_wmma<bf16, float><<<dim3(HID / 128, S_TOT / 128), 256, 0, stream>>>(
      aout, w_o, out, S_TOT, HID, NH * DH);
}
